// GATGENETAXONOMY_9431748182769
// MI455X (gfx1250) — matrix operand order repaired and checked by reading, not compiled, not run on hardware
//
#include <hip/hip_runtime.h>
#include <math.h>

// ---------------------------------------------------------------------------
// MI455X (gfx1250, wave32) implementation.
// Dense GEMMs -> V_WMMA_F32_16X16X32_F16. Block = 256 threads = 8 wave32s,
// covering a 32-row x 128-col output tile (each wave: two 16x16 D tiles).
// Tiles are staged in LDS in *fragment layout* (indexed by hardware lane),
// so fragment builds are ds_load_2addr_b64 pairs and global loads are
// float4 (global_load_b128). DIRECT GEMMs have K % 32 == 0 so the staging
// loops carry no bounds checks; only the 272-wide concat GEMM keeps a
// per-float4 guard. Bias is hoisted out of the epilogue loop (invariant
// per lane). Segment softmax uses order-preserving uint atomicMax; the
// edge-message GEMM fuses softmax-scale + scatter-add into the epilogue.
// ---------------------------------------------------------------------------

#define NN 100000
#define EE 400000
#define BG 4096
#define HD 256
#define GCOLS 768 /* 3*H for GRU gates */

typedef __attribute__((ext_vector_type(16))) _Float16 v16h;
typedef __attribute__((ext_vector_type(4)))  _Float16 v4h;
typedef __attribute__((ext_vector_type(8)))  float    v8f;

union F16Frag { v16h v; v4h q[4]; };

enum { LOADA_DIRECT = 0, LOADA_CONCAT = 2 };
enum { EPI_STORE = 0, EPI_SCATTER = 1 };
enum { ACT_NONE = 0, ACT_LEAKY = 1 };

__device__ __forceinline__ float leaky01(float v) { return v >= 0.f ? v : 0.01f * v; }

// order-preserving float<->uint for atomicMax-based segment max
__device__ __forceinline__ unsigned fenc(float f) {
  unsigned u = __float_as_uint(f);
  return (u & 0x80000000u) ? ~u : (u | 0x80000000u);
}
__device__ __forceinline__ float fdec(unsigned e) {
  unsigned u = (e & 0x80000000u) ? (e ^ 0x80000000u) : ~e;
  return __uint_as_float(u);
}
#define ENC_NEG_INF 0x007FFFFFu /* fenc(-inf) */

__device__ __forceinline__ v4h cvt4(float a, float b, float c, float d) {
  v4h h;
  h[0] = (_Float16)a; h[1] = (_Float16)b; h[2] = (_Float16)c; h[3] = (_Float16)d;
  return h;
}

// ---------------------------------------------------------------------------
// Tiled WMMA GEMM:  C[M,Nc] (+)= act( A[M,K] @ W[Nc,K]^T + bias )
//   LOADA_DIRECT : A row = A[row*K + k]           (requires K % 32 == 0)
//   LOADA_CONCAT : k<K1 from A[gidx[row]*K1+k], else A2[row*(K-K1)+(k-K1)]
//   EPI_STORE    : C[row*Nc + col] = act(acc + bias[col])
//   EPI_SCATTER  : atomicAdd(C[sidx[row]*Nc+col],
//                            acc * escale[row]/(ssum[sidx[row]]+1e-16))
// grid = (M/32, Nc/128), block = 256. M % 32 == 0, Nc % 128 == 0.
//
// LDS fragment layout: element i of hardware lane l lives at [l][i].
//   A (16x32 f16): lane l = m + 16*hi; i<8 -> K = hi*8+i; i>=8 -> K = 16+hi*8+(i-8)
//   B (32x16 f16): lane l = n + 16*hi; K from (hi, i) as for A
// Lane stride 20 f16 (40B): 8B-aligned v4h ops, conflict-free b64 banks.
// ---------------------------------------------------------------------------
template <int LOADA, int EPI, int ACT>
__global__ __launch_bounds__(256) void wmma_gemm(
    const float* __restrict__ A, const float* __restrict__ A2,
    const int* __restrict__ gidx, const float* __restrict__ W,
    const float* __restrict__ bias, float* __restrict__ C,
    const int* __restrict__ sidx, const float* __restrict__ escale,
    const float* __restrict__ ssum, int M, int K1, int K, int Nc) {
  (void)M;
  const int m0 = blockIdx.x * 32;
  const int wave = threadIdx.x >> 5;
  const int lane = threadIdx.x & 31;
  const int hi = lane >> 4;
  const int l15 = lane & 15;
  const int n0 = blockIdx.y * 128 + wave * 16;
  const int tid = threadIdx.x;

  __shared__ _Float16 Asf[2][32][20];   // two 16-row subtiles, fragment layout
  __shared__ _Float16 Bsf[8][32][20];   // per-wave B tile, fragment layout

  v8f acc0 = {}, acc1 = {};

  // A staging decomposition for this thread: one float4 per k-step
  const int ar = tid >> 3;          // 0..31 (row within 32-row tile)
  const int akk = (tid & 7) * 4;    // 0..28 (k within 32-k slice, 4-aligned)
  const int a_sub = ar >> 4;
  const int a_lane = (ar & 15) + 16 * ((akk >> 3) & 1);
  const int a_i = (akk & 7) + ((akk >> 4) << 3);

  for (int k0 = 0; k0 < K; k0 += 32) {
    // ---- stage A tile: 32 rows x 32 k, one float4 per thread ----
    {
      const int row = m0 + ar;
      const int gk = k0 + akk;
      v4h h4;
      if (LOADA == LOADA_DIRECT) {
        const float4 f = *(const float4*)&A[(long)row * K + gk];
        h4 = cvt4(f.x, f.y, f.z, f.w);
      } else {  // CONCAT (K1 % 4 == 0, K % 4 == 0; float4 never straddles)
        if (gk + 3 < K1) {
          const float4 f = *(const float4*)&A[(long)gidx[row] * K1 + gk];
          h4 = cvt4(f.x, f.y, f.z, f.w);
        } else if (gk + 3 < K) {
          const float4 f = *(const float4*)&A2[(long)row * (K - K1) + (gk - K1)];
          h4 = cvt4(f.x, f.y, f.z, f.w);
        } else {
          h4 = cvt4(0.f, 0.f, 0.f, 0.f);
        }
      }
      *(v4h*)&Asf[a_sub][a_lane][a_i] = h4;
    }
    // ---- stage B tile: 128 cols x 32 k, four float4 per thread ----
#pragma unroll
    for (int it = 0; it < 4; ++it) {
      const int idx = tid + it * 256;   // 0..1023
      const int nl = idx >> 3;          // 0..127
      const int kk = (idx & 7) * 4;     // 0..28
      const int gk = k0 + kk;
      const int gn = blockIdx.y * 128 + nl;
      v4h h4;
      if (LOADA == LOADA_DIRECT || gk + 3 < K) {
        const float4 f = *(const float4*)&W[(long)gn * K + gk];
        h4 = cvt4(f.x, f.y, f.z, f.w);
      } else {
        h4 = cvt4(0.f, 0.f, 0.f, 0.f);
      }
      const int b_lane = (nl & 15) + 16 * ((kk >> 3) & 1);
      *(v4h*)&Bsf[nl >> 4][b_lane][(kk & 7) + ((kk >> 4) << 3)] = h4;
    }
    __syncthreads();

    // ---- fragment loads (ds_load_2addr_b64 pairs) + two WMMAs ----
    F16Frag af0, af1, bf;
#pragma unroll
    for (int q = 0; q < 4; ++q) {
      af0.q[q] = *(const v4h*)&Asf[0][lane][q * 4];
      af1.q[q] = *(const v4h*)&Asf[1][lane][q * 4];
      bf.q[q]  = *(const v4h*)&Bsf[wave][lane][q * 4];
    }
    acc0 = __builtin_amdgcn_wmma_f32_16x16x32_f16(
        false, af0.v, false, bf.v, (short)0, acc0, false, false);
    acc1 = __builtin_amdgcn_wmma_f32_16x16x32_f16(
        false, af1.v, false, bf.v, (short)0, acc1, false, false);
    __syncthreads();
  }

  // ---- epilogue: D layout = {row = r + 8*hi, col = lane&15} ----
  const int col = n0 + l15;
  const float bv = (EPI == EPI_STORE && bias) ? bias[col] : 0.f;  // hoisted
#pragma unroll
  for (int s = 0; s < 2; ++s) {
    const v8f acc = s ? acc1 : acc0;
#pragma unroll
    for (int r = 0; r < 8; ++r) {
      const int row = m0 + s * 16 + hi * 8 + r;
      float v = acc[r] + bv;
      if (ACT == ACT_LEAKY) v = leaky01(v);
      if (EPI == EPI_STORE) {
        C[(long)row * Nc + col] = v;
      } else {
        const int d = sidx[row];
        const float sc = escale ? (escale[row] / (ssum[d] + 1e-16f)) : 1.f;
        atomicAdd(&C[(long)d * Nc + col], v * sc);
      }
    }
  }
}

// ---------------------------------------------------------------------------
// Small support kernels
// ---------------------------------------------------------------------------
__global__ void fill_f(float* p, float v, long n) {
  long t = (long)blockIdx.x * blockDim.x + threadIdx.x;
  if (t < n) p[t] = v;
}
__global__ void fill_u(unsigned* p, unsigned v, long n) {
  long t = (long)blockIdx.x * blockDim.x + threadIdx.x;
  if (t < n) p[t] = v;
}

// out[row] = dot(X[row,:K], v)   (one wave32 per row)
__global__ void rowdot(const float* __restrict__ X, const float* __restrict__ v,
                       float* __restrict__ out, int rows, int K) {
  int row = blockIdx.x * (blockDim.x >> 5) + (threadIdx.x >> 5);
  if (row >= rows) return;
  int lane = threadIdx.x & 31;
  float s = 0.f;
  for (int k = lane; k < K; k += 32) s += X[(long)row * K + k] * v[k];
  for (int off = 16; off; off >>= 1) s += __shfl_down(s, off, 32);
  if (lane == 0) out[row] = s;
}

// alpha[e] = leaky( pa[ia? ia[e]:e] + pb[ib[e]] )
__global__ void edge_alpha(const float* __restrict__ pa, const int* __restrict__ ia,
                           const float* __restrict__ pb, const int* __restrict__ ib,
                           float* __restrict__ alpha, int n) {
  int e = blockIdx.x * blockDim.x + threadIdx.x;
  if (e >= n) return;
  alpha[e] = leaky01(pa[ia ? ia[e] : e] + pb[ib[e]]);
}

__global__ void seg_amax(const float* __restrict__ alpha, const int* __restrict__ seg,
                         unsigned* __restrict__ mx, int n) {
  int e = blockIdx.x * blockDim.x + threadIdx.x;
  if (e >= n) return;
  atomicMax(&mx[seg[e]], fenc(alpha[e]));
}

// alpha := exp(alpha - max[seg]);  sum[seg] += alpha
__global__ void seg_expsum(float* __restrict__ alpha, const int* __restrict__ seg,
                           const unsigned* __restrict__ mx, float* __restrict__ sum,
                           int n) {
  int e = blockIdx.x * blockDim.x + threadIdx.x;
  if (e >= n) return;
  float m = fdec(mx[seg[e]]);
  if (!(m > -1e30f)) m = 0.f;  // matches where(isfinite(m), m, 0)
  float ev = expf(alpha[e] - m);
  alpha[e] = ev;
  atomicAdd(&sum[seg[e]], ev);
}

// out[gdst[row]] += X[gsrc? gsrc[row]:row] * escale[row]/(ssum[gdst]+eps)
__global__ void scatter_scale(const float* __restrict__ X, const int* __restrict__ gsrc,
                              const int* __restrict__ gdst,
                              const float* __restrict__ escale,
                              const float* __restrict__ ssum,
                              float* __restrict__ outp, long rows, int Kc) {
  long t = (long)blockIdx.x * blockDim.x + threadIdx.x;
  if (t >= rows * Kc) return;
  long row = t / Kc;
  int c = (int)(t - row * Kc);
  long sr = gsrc ? gsrc[row] : row;
  int d = gdst[row];
  float s = escale ? (escale[row] / (ssum[d] + 1e-16f)) : 1.f;
  atomicAdd(&outp[(long)d * Kc + c], X[sr * Kc + c] * s);
}

// y = act(y + bias[col]); act: 0 none, 1 relu, 2 elu
__global__ void bias_act(float* __restrict__ y, const float* __restrict__ bias,
                         int cols, long n, int act) {
  long t = (long)blockIdx.x * blockDim.x + threadIdx.x;
  if (t >= n) return;
  float v = y[t];
  if (bias) v += bias[(int)(t % cols)];
  if (act == 1) v = v > 0.f ? v : 0.f;
  else if (act == 2) v = v > 0.f ? v : (expf(v) - 1.f);
  y[t] = v;
}

// out = relu( (1-z)*n + z*hprev ), gates from gi/gh [rows,768]
__global__ void gru_ew(const float* __restrict__ gi, const float* __restrict__ gh,
                       const float* __restrict__ hprev, float* __restrict__ out,
                       long rows) {
  long t = (long)blockIdx.x * blockDim.x + threadIdx.x;
  if (t >= rows * HD) return;
  long row = t >> 8;
  int c = (int)(t & 255);
  float ir = gi[row * GCOLS + c], iz = gi[row * GCOLS + 256 + c], in = gi[row * GCOLS + 512 + c];
  float hr = gh[row * GCOLS + c], hz = gh[row * GCOLS + 256 + c], hn = gh[row * GCOLS + 512 + c];
  float r = 1.f / (1.f + expf(-(ir + hr)));
  float z = 1.f / (1.f + expf(-(iz + hz)));
  float nn = tanhf(in + r * hn);
  float v = (1.f - z) * nn + z * hprev[t];
  out[t] = v > 0.f ? v : 0.f;
}

// gene: Conv1d(4->1, k=3, s=3) then adaptive avg pool 1024 -> 256
__global__ void gene_kernel(const float* __restrict__ gene, const float* __restrict__ w,
                            const float* __restrict__ b, float* __restrict__ g, int Bq) {
  int t = blockIdx.x * blockDim.x + threadIdx.x;
  if (t >= Bq * HD) return;
  int bb = t >> 8, hc = t & 255;
  float accum = 0.f;
  for (int q = 0; q < 4; ++q) {
    int l = hc * 4 + q;
    float s = b[0];
    for (int c = 0; c < 4; ++c)
      for (int k = 0; k < 3; ++k)
        s += gene[((long)bb * 4 + c) * 3072 + l * 3 + k] * w[c * 3 + k];
    accum += s;
  }
  g[t] = accum * 0.25f;
}

__global__ void dur_kernel(const float* __restrict__ dur, const float* __restrict__ w,
                           const float* __restrict__ b, float* __restrict__ o, int Bq) {
  int t = blockIdx.x * blockDim.x + threadIdx.x;
  if (t >= Bq * HD) return;
  int bb = t >> 8, j = t & 255;
  float s = b[j];
  for (int k = 0; k < 8; ++k) s += dur[bb * 8 + k] * w[j * 8 + k];
  o[t] = s > 0.f ? s : 0.f;
}

__global__ void concat4(const float* a, const float* b, const float* c, const float* d,
                        float* o, int Bq) {
  int t = blockIdx.x * blockDim.x + threadIdx.x;
  if (t >= Bq * 1024) return;
  int bb = t >> 10, col = t & 1023;
  float v;
  if (col < 256)       v = a[bb * 256 + col];
  else if (col < 512)  v = b[bb * 256 + col - 256];
  else if (col < 768)  v = c[bb * 256 + col - 512];
  else                 v = d[bb * 256 + col - 768];
  o[t] = v;
}

__global__ void lin5_kernel(const float* __restrict__ x, const float* __restrict__ w,
                            const float* __restrict__ b, float* __restrict__ o, int Bq) {
  int row = blockIdx.x * (blockDim.x >> 5) + (threadIdx.x >> 5);
  if (row >= Bq) return;
  int lane = threadIdx.x & 31;
  float s = 0.f;
  for (int k = lane; k < 256; k += 32) s += x[(long)row * 256 + k] * w[k];
  for (int off = 16; off; off >>= 1) s += __shfl_down(s, off, 32);
  if (lane == 0) o[row] = s + b[0];
}

// ---------------------------------------------------------------------------
extern "C" void kernel_launch(void* const* d_in, const int* in_sizes, int n_in,
                              void* d_out, int out_size, void* d_ws, size_t ws_size,
                              hipStream_t stream) {
  (void)in_sizes; (void)n_in; (void)out_size; (void)ws_size;
  const long Nn = NN, Ee = EE, Bq = BG;

  const float* x     = (const float*)d_in[0];
  const int*   eidx  = (const int*)d_in[1];
  const int*   src   = eidx;
  const int*   dst   = eidx + Ee;
  const float* eattr = (const float*)d_in[2];
  const int*   batch = (const int*)d_in[3];
  const float* gene  = (const float*)d_in[4];
  const float* tax   = (const float*)d_in[5];
  const float* dur   = (const float*)d_in[6];
#define P(i) ((const float*)d_in[i])
  const float *lin1_w = P(7), *lin1_b = P(8);
  const float *g1w = P(9), *g2w = P(10), *gattl = P(11), *gattr = P(12), *gbias = P(13);
  const float *gru0_wih = P(14), *gru0_whh = P(15), *gru0_bih = P(16), *gru0_bhh = P(17);
  const float* convW[2]  = {P(18), P(22)};
  const float* convAs[2] = {P(19), P(23)};
  const float* convAd[2] = {P(20), P(24)};
  const float* convB[2]  = {P(21), P(25)};
  const float* gwih[2] = {P(26), P(30)};
  const float* gwhh[2] = {P(27), P(31)};
  const float* gbih[2] = {P(28), P(32)};
  const float* gbhh[2] = {P(29), P(33)};
  const float *molW = P(34), *molAs = P(35), *molAd = P(36), *molB = P(37);
  const float *mg_wih = P(38), *mg_whh = P(39), *mg_bih = P(40), *mg_bhh = P(41);
  const float *genew = P(42), *geneb = P(43);
  const float *durw = P(44), *durb = P(45);
  const float *l4w = P(46), *l4b = P(47), *l5w = P(48), *l5b = P(49);
#undef P

  // ---- workspace carve-out (~1.0 GB of f32) ----
  float* ws = (float*)d_ws;
  long o = 0;
  float* xh   = ws + o; o += Nn * 256;   // node state
  float* hbuf = ws + o; o += Nn * 256;   // conv candidate
  float* xwb  = ws + o; o += Nn * 256;   // xw / xw_src
  float* gi   = ws + o; o += Nn * 768;   // GRU input gates (aliases edge m)
  float* gh   = ws + o; o += Nn * 768;   // GRU hidden gates
  float* mbuf = gi;                      // edge m [E,256] fits in gi+gh region
  float* alpha = ws + o; o += Ee;
  float* dots  = ws + o; o += Ee;        // m@att_l / a_src(node)
  float* ndot  = ws + o; o += Nn;        // dst-side dots
  unsigned* nmax = (unsigned*)(ws + o); o += Nn;
  float* nsum = ws + o; o += Nn;
  float* outB = ws + o; o += Bq * 256;
  float* hB   = ws + o; o += Bq * 256;
  float* owB  = ws + o; o += Bq * 256;
  float* giB  = ws + o; o += Bq * 768;
  float* ghB  = ws + o; o += Bq * 768;
  float* gbuf = ws + o; o += Bq * 256;
  float* durB = ws + o; o += Bq * 256;
  float* catB = ws + o; o += Bq * 1024;
  float* l4B  = ws + o; o += Bq * 256;

  auto cdiv = [](long a, long b) { return (unsigned)((a + b - 1) / b); };
  const dim3 blk(256);

  // ---- 1. lin1: xh = leaky(x @ lin1_w.T + b) ----
  wmma_gemm<LOADA_DIRECT, EPI_STORE, ACT_LEAKY><<<dim3(Nn / 32, 2), blk, 0, stream>>>(
      x, nullptr, nullptr, lin1_w, lin1_b, xh, nullptr, nullptr, nullptr,
      (int)Nn, 64, 64, 256);

  // ---- 2. GATEConv ----
  // m = leaky(concat(xh[src], edge_attr) @ g1w.T)   [E,256], K=272
  wmma_gemm<LOADA_CONCAT, EPI_STORE, ACT_LEAKY><<<dim3(Ee / 32, 2), blk, 0, stream>>>(
      xh, eattr, src, g1w, nullptr, mbuf, nullptr, nullptr, nullptr,
      (int)Ee, 256, 272, 256);
  rowdot<<<cdiv(Ee, 8), blk, 0, stream>>>(mbuf, gattl, dots, (int)Ee, 256);
  rowdot<<<cdiv(Nn, 8), blk, 0, stream>>>(xh, gattr, ndot, (int)Nn, 256);
  edge_alpha<<<cdiv(Ee, 256), blk, 0, stream>>>(dots, nullptr, ndot, dst, alpha, (int)Ee);
  fill_u<<<cdiv(Nn, 256), blk, 0, stream>>>(nmax, ENC_NEG_INF, Nn);
  seg_amax<<<cdiv(Ee, 256), blk, 0, stream>>>(alpha, dst, nmax, (int)Ee);
  fill_f<<<cdiv(Nn, 256), blk, 0, stream>>>(nsum, 0.f, Nn);
  seg_expsum<<<cdiv(Ee, 256), blk, 0, stream>>>(alpha, dst, nmax, nsum, (int)Ee);
  fill_f<<<cdiv(Nn * 256, 256), blk, 0, stream>>>(hbuf, 0.f, Nn * 256);
  // h[dst] += (m @ g2w.T) * softmax  — fused into WMMA epilogue
  wmma_gemm<LOADA_DIRECT, EPI_SCATTER, ACT_NONE><<<dim3(Ee / 32, 2), blk, 0, stream>>>(
      mbuf, nullptr, nullptr, g2w, nullptr, hbuf, dst, alpha, nsum,
      (int)Ee, 256, 256, 256);
  bias_act<<<cdiv(Nn * 256, 256), blk, 0, stream>>>(hbuf, gbias, 256, Nn * 256, 2 /*elu*/);
  // xh = relu(gru0(hbuf, xh))
  wmma_gemm<LOADA_DIRECT, EPI_STORE, ACT_NONE><<<dim3(Nn / 32, 6), blk, 0, stream>>>(
      hbuf, nullptr, nullptr, gru0_wih, gru0_bih, gi, nullptr, nullptr, nullptr,
      (int)Nn, 256, 256, 768);
  wmma_gemm<LOADA_DIRECT, EPI_STORE, ACT_NONE><<<dim3(Nn / 32, 6), blk, 0, stream>>>(
      xh, nullptr, nullptr, gru0_whh, gru0_bhh, gh, nullptr, nullptr, nullptr,
      (int)Nn, 256, 256, 768);
  gru_ew<<<cdiv(Nn * 256, 256), blk, 0, stream>>>(gi, gh, xh, xh, Nn);

  // ---- 3. atom GATConv layers ----
  for (int l = 0; l < 2; ++l) {
    wmma_gemm<LOADA_DIRECT, EPI_STORE, ACT_NONE><<<dim3(Nn / 32, 2), blk, 0, stream>>>(
        xh, nullptr, nullptr, convW[l], nullptr, xwb, nullptr, nullptr, nullptr,
        (int)Nn, 256, 256, 256);
    rowdot<<<cdiv(Nn, 8), blk, 0, stream>>>(xwb, convAs[l], dots, (int)Nn, 256);
    rowdot<<<cdiv(Nn, 8), blk, 0, stream>>>(xwb, convAd[l], ndot, (int)Nn, 256);
    edge_alpha<<<cdiv(Ee, 256), blk, 0, stream>>>(dots, src, ndot, dst, alpha, (int)Ee);
    fill_u<<<cdiv(Nn, 256), blk, 0, stream>>>(nmax, ENC_NEG_INF, Nn);
    seg_amax<<<cdiv(Ee, 256), blk, 0, stream>>>(alpha, dst, nmax, (int)Ee);
    fill_f<<<cdiv(Nn, 256), blk, 0, stream>>>(nsum, 0.f, Nn);
    seg_expsum<<<cdiv(Ee, 256), blk, 0, stream>>>(alpha, dst, nmax, nsum, (int)Ee);
    fill_f<<<cdiv(Nn * 256, 256), blk, 0, stream>>>(hbuf, 0.f, Nn * 256);
    scatter_scale<<<cdiv(Ee * 256, 256), blk, 0, stream>>>(
        xwb, src, dst, alpha, nsum, hbuf, Ee, 256);
    bias_act<<<cdiv(Nn * 256, 256), blk, 0, stream>>>(hbuf, convB[l], 256, Nn * 256, 2);
    wmma_gemm<LOADA_DIRECT, EPI_STORE, ACT_NONE><<<dim3(Nn / 32, 6), blk, 0, stream>>>(
        hbuf, nullptr, nullptr, gwih[l], gbih[l], gi, nullptr, nullptr, nullptr,
        (int)Nn, 256, 256, 768);
    wmma_gemm<LOADA_DIRECT, EPI_STORE, ACT_NONE><<<dim3(Nn / 32, 6), blk, 0, stream>>>(
        xh, nullptr, nullptr, gwhh[l], gbhh[l], gh, nullptr, nullptr, nullptr,
        (int)Nn, 256, 256, 768);
    gru_ew<<<cdiv(Nn * 256, 256), blk, 0, stream>>>(gi, gh, xh, xh, Nn);
  }

  // ---- 4. molecule readout (bipartite GAT, NUM_T=2) ----
  fill_f<<<cdiv((long)Bq * 256, 256), blk, 0, stream>>>(outB, 0.f, (long)Bq * 256);
  scatter_scale<<<cdiv(Nn * 256, 256), blk, 0, stream>>>(
      xh, nullptr, batch, nullptr, nullptr, outB, Nn, 256);
  bias_act<<<cdiv((long)Bq * 256, 256), blk, 0, stream>>>(outB, nullptr, 256,
                                                          (long)Bq * 256, 1 /*relu*/);
  wmma_gemm<LOADA_DIRECT, EPI_STORE, ACT_NONE><<<dim3(Nn / 32, 2), blk, 0, stream>>>(
      xh, nullptr, nullptr, molW, nullptr, xwb, nullptr, nullptr, nullptr,
      (int)Nn, 256, 256, 256);                       // xw_src (constant over t)
  rowdot<<<cdiv(Nn, 8), blk, 0, stream>>>(xwb, molAs, dots, (int)Nn, 256);  // a_src
  for (int t = 0; t < 2; ++t) {
    wmma_gemm<LOADA_DIRECT, EPI_STORE, ACT_NONE><<<dim3(Bq / 32, 2), blk, 0, stream>>>(
        outB, nullptr, nullptr, molW, nullptr, owB, nullptr, nullptr, nullptr,
        (int)Bq, 256, 256, 256);
    rowdot<<<cdiv(Bq, 8), blk, 0, stream>>>(owB, molAd, ndot, (int)Bq, 256);
    edge_alpha<<<cdiv(Nn, 256), blk, 0, stream>>>(dots, nullptr, ndot, batch, alpha, (int)Nn);
    fill_u<<<cdiv(Bq, 256), blk, 0, stream>>>(nmax, ENC_NEG_INF, Bq);
    seg_amax<<<cdiv(Nn, 256), blk, 0, stream>>>(alpha, batch, nmax, (int)Nn);
    fill_f<<<cdiv(Bq, 256), blk, 0, stream>>>(nsum, 0.f, Bq);
    seg_expsum<<<cdiv(Nn, 256), blk, 0, stream>>>(alpha, batch, nmax, nsum, (int)Nn);
    fill_f<<<cdiv((long)Bq * 256, 256), blk, 0, stream>>>(hB, 0.f, (long)Bq * 256);
    scatter_scale<<<cdiv(Nn * 256, 256), blk, 0, stream>>>(
        xwb, nullptr, batch, alpha, nsum, hB, Nn, 256);
    bias_act<<<cdiv((long)Bq * 256, 256), blk, 0, stream>>>(hB, molB, 256,
                                                            (long)Bq * 256, 2);
    wmma_gemm<LOADA_DIRECT, EPI_STORE, ACT_NONE><<<dim3(Bq / 32, 6), blk, 0, stream>>>(
        hB, nullptr, nullptr, mg_wih, mg_bih, giB, nullptr, nullptr, nullptr,
        (int)Bq, 256, 256, 768);
    wmma_gemm<LOADA_DIRECT, EPI_STORE, ACT_NONE><<<dim3(Bq / 32, 6), blk, 0, stream>>>(
        outB, nullptr, nullptr, mg_whh, mg_bhh, ghB, nullptr, nullptr, nullptr,
        (int)Bq, 256, 256, 768);
    gru_ew<<<cdiv((long)Bq * 256, 256), blk, 0, stream>>>(giB, ghB, outB, outB, Bq);
  }

  // ---- 5. gene / duration / concat / lin4 / lin5 ----
  gene_kernel<<<cdiv((long)Bq * 256, 256), blk, 0, stream>>>(gene, genew, geneb, gbuf, (int)Bq);
  dur_kernel<<<cdiv((long)Bq * 256, 256), blk, 0, stream>>>(dur, durw, durb, durB, (int)Bq);
  concat4<<<cdiv((long)Bq * 1024, 256), blk, 0, stream>>>(outB, gbuf, tax, durB, catB, (int)Bq);
  wmma_gemm<LOADA_DIRECT, EPI_STORE, ACT_NONE><<<dim3(Bq / 32, 2), blk, 0, stream>>>(
      catB, nullptr, nullptr, l4w, l4b, l4B, nullptr, nullptr, nullptr,
      (int)Bq, 1024, 1024, 256);
  lin5_kernel<<<cdiv(Bq, 8), blk, 0, stream>>>(l4B, l5w, l5b, (float*)d_out, (int)Bq);
}
